// CausalSelfAttention_25632364822759
// MI455X (gfx1250) — compile-verified, matrix operand order checked by reading, not run on hardware
//
#include <hip/hip_runtime.h>

typedef unsigned short u16t;
typedef __attribute__((ext_vector_type(16))) __bf16 bf16x16;
typedef __attribute__((ext_vector_type(8)))  float  f32x8;

#define WMMA_BF16(a,b,c) \
  __builtin_amdgcn_wmma_f32_16x16x32_bf16(false,(a),false,(b),(short)0,(c),false,false)

constexpr int Bb = 4, Tt = 2048, Cch = 1024, Hh = 16, Dd = 64;
constexpr int Mrows = Bb * Tt; // 8192

// ---- helpers -------------------------------------------------------------

__device__ __forceinline__ u16t f2bf(float f) {
  unsigned u = __float_as_uint(f);
  u += 0x7fffu + ((u >> 16) & 1u);       // round-to-nearest-even
  return (u16t)(u >> 16);
}

// Load one 16-element bf16 fragment as two contiguous 8-element (16B) runs.
// ISA 16-bit A/B fragment layout: element e -> K = 8g+e (e<8), 16+8g+(e-8),
// with g = lane/16.
__device__ __forceinline__ bf16x16 load_frag(const u16t* p0, const u16t* p1) {
  union { bf16x16 v; uint4 q[2]; } u;
  u.q[0] = *(const uint4*)p0;
  u.q[1] = *(const uint4*)p1;
  return u.v;
}

// ---- conversion kernels --------------------------------------------------

__global__ void cvt_x_kernel(const float* __restrict__ x, u16t* __restrict__ o) {
  size_t id = (size_t)blockIdx.x * blockDim.x + threadIdx.x;
  float4 f = ((const float4*)x)[id];
  union { u16t s[4]; uint2 u; } p;
  p.s[0] = f2bf(f.x); p.s[1] = f2bf(f.y);
  p.s[2] = f2bf(f.z); p.s[3] = f2bf(f.w);
  ((uint2*)o)[id] = p.u;
}

__global__ void cvt_wT_kernel(const float* __restrict__ W, u16t* __restrict__ Wt) {
  int id = blockIdx.x * blockDim.x + threadIdx.x;   // id = k*1024 + n
  int k = id >> 10, n = id & 1023;
  Wt[(size_t)n * Cch + k] = f2bf(W[id]);
}

// ---- GEMM: one wave = 32x64 output tile (spill-free register budget) -----
// Y = A(M x 1024, bf16) @ W(1024 x 1024) + bias.  Wt is W transposed (N x K).
// mode: 0 = bias+RoPE -> [b,h,t,d] bf16
//       1 = bias -> V transposed [b,h,d,t] bf16
//       2 = bias -> f32 row-major output

__global__ __launch_bounds__(256) void gemm_kernel(
    const u16t* __restrict__ A,
    const u16t* __restrict__ Wt,
    const float* __restrict__ bias,
    void* __restrict__ dst, int mode) {
  int lane = threadIdx.x & 31, wid = threadIdx.x >> 5;
  int wg = blockIdx.x * 8 + wid;            // 4096 waves total
  int m0 = (wg >> 4) * 32;                  // 256 row tiles of 32
  int n0 = (wg & 15) * 64;                  // 16 col tiles of 64
  int lane16 = lane & 15, g = lane >> 4;

  const u16t* arow = A  + (size_t)(m0 + lane16) * Cch;
  const u16t* brow = Wt + (size_t)(n0 + lane16) * Cch;

  const f32x8 zero = {0,0,0,0,0,0,0,0};
  f32x8 acc[8];
#pragma unroll
  for (int i = 0; i < 8; ++i) acc[i] = zero;

  for (int kc = 0; kc < Cch; kc += 32) {
    bf16x16 bf[4];
#pragma unroll
    for (int j = 0; j < 4; ++j) {
      const u16t* bp = brow + (size_t)j * 16 * Cch + kc + 8 * g;
      bf[j] = load_frag(bp, bp + 16);
    }
#pragma unroll
    for (int i = 0; i < 2; ++i) {
      const u16t* ap = arow + (size_t)i * 16 * Cch + kc + 8 * g;
      bf16x16 af = load_frag(ap, ap + 16);
      acc[i * 4 + 0] = WMMA_BF16(af, bf[0], acc[i * 4 + 0]);
      acc[i * 4 + 1] = WMMA_BF16(af, bf[1], acc[i * 4 + 1]);
      acc[i * 4 + 2] = WMMA_BF16(af, bf[2], acc[i * 4 + 2]);
      acc[i * 4 + 3] = WMMA_BF16(af, bf[3], acc[i * 4 + 3]);
    }
  }

#pragma unroll
  for (int j = 0; j < 4; ++j) {
    int colg = n0 + j * 16 + lane16;
    float bv = bias[colg];
    int h = colg >> 6, d = colg & 63;
    float freq = __expf(-0.29710775393f * (float)(d >> 1)); // ln(1e4)/31
#pragma unroll
    for (int i = 0; i < 2; ++i) {
#pragma unroll
      for (int r = 0; r < 8; ++r) {
        int rowg = m0 + i * 16 + r + 8 * g;   // C-layout: M = r + 8*(lane/16)
        float val = acc[i * 4 + j][r] + bv;
        if (mode == 2) {
          ((float*)dst)[(size_t)rowg * Cch + colg] = val;
        } else if (mode == 1) {
          int b = rowg >> 11, t = rowg & (Tt - 1);
          ((u16t*)dst)[((size_t)(b * Hh + h) * Dd + d) * Tt + t] = f2bf(val);
        } else {
          int b = rowg >> 11, t = rowg & (Tt - 1);
          float ang = (float)t * freq;
          float sn, cs; __sincosf(ang, &sn, &cs);
          float other = __shfl_xor(val, 1);   // rotary partner (even,odd)
          float outv = (d & 1) ? (other * sn + val * cs)   // xi' = xr*s + xi*c
                               : (val * cs - other * sn);  // xr' = xr*c - xi*s
          ((u16t*)dst)[((size_t)(b * Hh + h) * Tt + t) * Dd + d] = f2bf(outv);
        }
      }
    }
  }
}

// ---- Flash attention: one wave = one (b,h) x 16-query block --------------
// (16 queries keeps live VGPRs ~130 -> no scratch spills; K/V stream from L2.)

__global__ __launch_bounds__(256) void attn_kernel(
    const u16t* __restrict__ qg,
    const u16t* __restrict__ kg,
    const u16t* __restrict__ vt,
    u16t* __restrict__ att) {
  __shared__ u16t ptile[8][16 * 32];        // per-wave P tile (bf16)
  int lane = threadIdx.x & 31, wid = threadIdx.x >> 5;
  int wg = blockIdx.x * 8 + wid;            // 8192 waves
  int qb = wg & 127;                        // query block (T/16)
  int bh = wg >> 7;                         // b*H + h, 0..63
  int lane16 = lane & 15, g = lane >> 4;

  const f32x8 zero = {0,0,0,0,0,0,0,0};

  // Q fragments for d = 0..31 and d = 32..63
  const u16t* qbase = qg + ((size_t)bh * Tt + qb * 16 + lane16) * Dd;
  bf16x16 qf0 = load_frag(qbase +      8 * g, qbase + 16 + 8 * g);
  bf16x16 qf1 = load_frag(qbase + 32 + 8 * g, qbase + 48 + 8 * g);

  f32x8 o0 = zero, o1 = zero, o2 = zero, o3 = zero;
  float m[8], l[8];
#pragma unroll
  for (int r = 0; r < 8; ++r) { m[r] = -1e30f; l[r] = 0.0f; }

  int smax = qb * 16 + 16;                  // keys needed (causal)
  for (int s0 = 0; s0 < smax; s0 += 32) {
    // K fragments: keys [s0, s0+16) and [s0+16, s0+32)
    const u16t* kb0 = kg + ((size_t)bh * Tt + s0 + lane16) * Dd;
    const u16t* kb1 = kb0 + 16 * Dd;
    bf16x16 k00 = load_frag(kb0 +      8 * g, kb0 + 16 + 8 * g);
    bf16x16 k01 = load_frag(kb0 + 32 + 8 * g, kb0 + 48 + 8 * g);
    bf16x16 k10 = load_frag(kb1 +      8 * g, kb1 + 16 + 8 * g);
    bf16x16 k11 = load_frag(kb1 + 32 + 8 * g, kb1 + 48 + 8 * g);

    f32x8 sA = zero; sA = WMMA_BF16(qf0, k00, sA); sA = WMMA_BF16(qf1, k01, sA);
    f32x8 sB = zero; sB = WMMA_BF16(qf0, k10, sB); sB = WMMA_BF16(qf1, k11, sB);

    float p0[8], p1[8];
#pragma unroll
    for (int r = 0; r < 8; ++r) {
      int t = qb * 16 + r + 8 * g;
      float v0 = sA[r] * 0.125f;            // 1/sqrt(64)
      float v1 = sB[r] * 0.125f;
      if (s0 +      lane16 > t) v0 = -1e30f;
      if (s0 + 16 + lane16 > t) v1 = -1e30f;
      float cur = fmaxf(v0, v1);
      cur = fmaxf(cur, __shfl_xor(cur, 1));
      cur = fmaxf(cur, __shfl_xor(cur, 2));
      cur = fmaxf(cur, __shfl_xor(cur, 4));
      cur = fmaxf(cur, __shfl_xor(cur, 8));  // row max over 32 keys
      float mn = fmaxf(m[r], cur);
      float alpha = __expf(m[r] - mn);
      float e0 = __expf(v0 - mn), e1 = __expf(v1 - mn);
      float rs = e0 + e1;
      rs += __shfl_xor(rs, 1);
      rs += __shfl_xor(rs, 2);
      rs += __shfl_xor(rs, 4);
      rs += __shfl_xor(rs, 8);               // row sum over 32 keys
      l[r] = l[r] * alpha + rs;
      m[r] = mn;
      o0[r] *= alpha; o1[r] *= alpha; o2[r] *= alpha; o3[r] *= alpha;
      p0[r] = e0; p1[r] = e1;
    }

    // C-layout -> A-fragment conversion via the per-wave LDS tile (same-wave
    // DS ops are in order; compiler inserts s_wait_dscnt for the reload).
    u16t* pt = ptile[wid];
#pragma unroll
    for (int r = 0; r < 8; ++r) {
      pt[(r + 8 * g) * 32 +      lane16] = f2bf(p0[r]);
      pt[(r + 8 * g) * 32 + 16 + lane16] = f2bf(p1[r]);
    }
    const u16t* pr = pt + lane16 * 32;
    bf16x16 pf = load_frag(pr + 8 * g, pr + 16 + 8 * g);

    // PV: B fragment from transposed V, runs contiguous along s
    const u16t* vb = vt + ((size_t)bh * Dd + lane16) * Tt + s0;
    o0 = WMMA_BF16(pf, load_frag(vb + 8 * g, vb + 16 + 8 * g), o0); vb += 16 * Tt;
    o1 = WMMA_BF16(pf, load_frag(vb + 8 * g, vb + 16 + 8 * g), o1); vb += 16 * Tt;
    o2 = WMMA_BF16(pf, load_frag(vb + 8 * g, vb + 16 + 8 * g), o2); vb += 16 * Tt;
    o3 = WMMA_BF16(pf, load_frag(vb + 8 * g, vb + 16 + 8 * g), o3);
  }

  // Normalize and store to att[b*T+t][h*64+d] (bf16) for the final GEMM.
  int b = bh >> 4, h = bh & 15;
#pragma unroll
  for (int r = 0; r < 8; ++r) {
    float inv = 1.0f / l[r];
    int t = qb * 16 + r + 8 * g;
    size_t row = ((size_t)b * Tt + t) * Cch + h * 64 + lane16;
    att[row +  0] = f2bf(o0[r] * inv);
    att[row + 16] = f2bf(o1[r] * inv);
    att[row + 32] = f2bf(o2[r] * inv);
    att[row + 48] = f2bf(o3[r] * inv);
  }
}

// ---- launch --------------------------------------------------------------
// Workspace layout (bytes): x_bf 16.8M | 4x Wt 2.1M | q 16.8M | k 16.8M |
// vT 16.8M | att 16.8M  => ~92.3 MB total.

extern "C" void kernel_launch(void* const* d_in, const int* in_sizes, int n_in,
                              void* d_out, int out_size, void* d_ws, size_t ws_size,
                              hipStream_t stream) {
  const float* x  = (const float*)d_in[0];
  const float* Wq = (const float*)d_in[1];
  const float* bq = (const float*)d_in[2];
  const float* Wk = (const float*)d_in[3];
  const float* bk = (const float*)d_in[4];
  const float* Wv = (const float*)d_in[5];
  const float* bv = (const float*)d_in[6];
  const float* Wp = (const float*)d_in[7];
  const float* bp = (const float*)d_in[8];
  float* out = (float*)d_out;

  u16t* xbf = (u16t*)d_ws;
  u16t* wtq = xbf + (size_t)Mrows * Cch;
  u16t* wtk = wtq + (size_t)Cch * Cch;
  u16t* wtv = wtk + (size_t)Cch * Cch;
  u16t* wtp = wtv + (size_t)Cch * Cch;
  u16t* qbf = wtp + (size_t)Cch * Cch;
  u16t* kbf = qbf + (size_t)Mrows * Cch;
  u16t* vtb = kbf + (size_t)Mrows * Cch;
  u16t* atb = vtb + (size_t)Mrows * Cch;

  cvt_x_kernel<<<Mrows * Cch / 1024, 256, 0, stream>>>(x, xbf);
  cvt_wT_kernel<<<Cch * Cch / 256, 256, 0, stream>>>(Wq, wtq);
  cvt_wT_kernel<<<Cch * Cch / 256, 256, 0, stream>>>(Wk, wtk);
  cvt_wT_kernel<<<Cch * Cch / 256, 256, 0, stream>>>(Wv, wtv);
  cvt_wT_kernel<<<Cch * Cch / 256, 256, 0, stream>>>(Wp, wtp);

  gemm_kernel<<<512, 256, 0, stream>>>(xbf, wtq, bq, (void*)qbf, 0); // Q + RoPE
  gemm_kernel<<<512, 256, 0, stream>>>(xbf, wtk, bk, (void*)kbf, 0); // K + RoPE
  gemm_kernel<<<512, 256, 0, stream>>>(xbf, wtv, bv, (void*)vtb, 1); // V (transposed)

  attn_kernel<<<1024, 256, 0, stream>>>(qbf, kbf, vtb, atb);

  gemm_kernel<<<512, 256, 0, stream>>>(atb, wtp, bp, (void*)out, 2); // projection
}
